// MuOrStdFromDistribution_46325517254762
// MI455X (gfx1250) — matrix operand order repaired and checked by reading, not compiled, not run on hardware
//
#include <hip/hip_runtime.h>

// MI455X / gfx1250: wave32, WMMA 16x16x32 bf16 (fp32 accum).
// One-pass Gram S = x^T x + column sums via bf16x2-split WMMA with symmetric
// reconstruction:  Q = H^T H + 2 H^T L  accumulated per tile (2 WMMAs/tile),
// S = 0.5 (Q + Q^T) applied in the atomic epilogue. Then a tiny finalize
// kernel computes cov = S/N - mu mu^T and the head network.

typedef __attribute__((ext_vector_type(16))) __bf16 v16bf;
typedef __attribute__((ext_vector_type(8)))  float  v8f;

#define CDIM 128
#define HDIM 3
#define PITCH 34              // ushorts per LDS row (68 B: 17-bank stride, conflict-friendly)
#define PLANE (CDIM * PITCH)  // 4352 ushorts per plane (hi / 2*lo)

__device__ __forceinline__ unsigned short f2bf_rne(float f) {
  union { float f; unsigned int u; } x; x.f = f;
  unsigned int r = x.u + 0x7fffu + ((x.u >> 16) & 1u);
  return (unsigned short)(r >> 16);
}
__device__ __forceinline__ float bf2f(unsigned short h) {
  union { unsigned int u; float f; } x; x.u = ((unsigned int)h) << 16;
  return x.f;
}
// Convert two fp32 (consecutive rows n, n+1) into packed bf16-hi and bf16-(2*lo).
__device__ __forceinline__ void pack2(float f0, float f1,
                                      unsigned int& hi, unsigned int& lo2) {
  unsigned short h0 = f2bf_rne(f0), h1 = f2bf_rne(f1);
  unsigned short l0 = f2bf_rne(2.0f * (f0 - bf2f(h0)));
  unsigned short l1 = f2bf_rne(2.0f * (f1 - bf2f(h1)));
  hi  = (unsigned int)h0 | ((unsigned int)h1 << 16);
  lo2 = (unsigned int)l0 | ((unsigned int)l1 << 16);
}

union FragU { v16bf v; unsigned int u[8]; };

__global__ void __launch_bounds__(256)
gram_colsum_kernel(const float* __restrict__ x, float* __restrict__ S,
                   float* __restrict__ colsum, int N) {
  // Transposed staging: ldsT[plane][c][n], c = 0..127 columns, n = 0..31 chunk rows.
  __shared__ unsigned short lds[2 * PLANE];

  const int tid  = threadIdx.x;
  const int lane = tid & 31;
  const int wv   = tid >> 5;           // wave id 0..7
  const int rp   = wv >> 1;            // tile-row pair: rows 2rp, 2rp+1
  const int chh  = wv & 1;             // tile-col half: cols 4chh .. 4chh+3
  const int cg   = (tid & 31) * 4;     // this thread's 4 columns (fixed across chunks)
  const int m    = lane & 15;
  const int half = lane >> 4;

  float cs0 = 0.f, cs1 = 0.f, cs2 = 0.f, cs3 = 0.f;
  v8f acc[2][4] = {};                  // Q tiles: (row 2rp+r, col 4chh+j)

  const int nch = N >> 5;              // N = 1e6 divisible by 32
  for (int ch = blockIdx.x; ch < nch; ch += gridDim.x) {
    __syncthreads();                   // protect LDS from previous compute readers

    // ---- stage 32x128 fp32 chunk -> transposed bf16 hi / 2*lo planes ----
    const float* xb = x + (size_t)ch * 32 * CDIM;
#pragma unroll
    for (int p = 0; p < 2; ++p) {
      const int n0 = (p * 8 + wv) * 2;          // even row within chunk
      const float4 va = *(const float4*)(xb + (size_t)n0 * CDIM + cg);
      const float4 vb = *(const float4*)(xb + (size_t)(n0 + 1) * CDIM + cg);
      unsigned int ph, pl;
      pack2(va.x, vb.x, ph, pl); cs0 += va.x + vb.x;
      *(unsigned int*)(lds + (cg + 0) * PITCH + n0) = ph;
      *(unsigned int*)(lds + PLANE + (cg + 0) * PITCH + n0) = pl;
      pack2(va.y, vb.y, ph, pl); cs1 += va.y + vb.y;
      *(unsigned int*)(lds + (cg + 1) * PITCH + n0) = ph;
      *(unsigned int*)(lds + PLANE + (cg + 1) * PITCH + n0) = pl;
      pack2(va.z, vb.z, ph, pl); cs2 += va.z + vb.z;
      *(unsigned int*)(lds + (cg + 2) * PITCH + n0) = ph;
      *(unsigned int*)(lds + PLANE + (cg + 2) * PITCH + n0) = pl;
      pack2(va.w, vb.w, ph, pl); cs3 += va.w + vb.w;
      *(unsigned int*)(lds + (cg + 3) * PITCH + n0) = ph;
      *(unsigned int*)(lds + PLANE + (cg + 3) * PITCH + n0) = pl;
    }
    __syncthreads();

    // ---- A-hi fragments (16x32 bf16) for tile rows 2rp, 2rp+1 ----
    FragU a[2];
#pragma unroll
    for (int r = 0; r < 2; ++r) {
      const unsigned int* ra =
          (const unsigned int*)(lds + ((2 * rp + r) * 16 + m) * PITCH);
#pragma unroll
      for (int i = 0; i < 4; ++i) {
        a[r].u[i]     = ra[half * 4 + i];      // K = 8*half + 0..7
        a[r].u[4 + i] = ra[8 + half * 4 + i];  // K = 16 + 8*half + 0..7
      }
    }
    // ---- B fragments (32x16 bf16) for tile cols 4chh..4chh+3, hi and 2*lo ----
    FragU bhi[4], blo[4];
#pragma unroll
    for (int j = 0; j < 4; ++j) {
      const int jj = 4 * chh + j;
      const unsigned int* r0 = (const unsigned int*)(lds + (jj * 16 + m) * PITCH);
      const unsigned int* r1 = (const unsigned int*)(lds + PLANE + (jj * 16 + m) * PITCH);
#pragma unroll
      for (int i = 0; i < 8; ++i) {
        bhi[j].u[i] = r0[half * 4 + i + (i & 4)];  // K = 8*half + 0..7, 16 + 8*half + 0..7, N = lane&15
        blo[j].u[i] = r1[half * 4 + i + (i & 4)];
      }
    }
    // ---- 16 back-to-back WMMAs: Q += H^T H + 2 H^T L ----
#pragma unroll
    for (int r = 0; r < 2; ++r) {
#pragma unroll
      for (int j = 0; j < 4; ++j) {
        acc[r][j] = __builtin_amdgcn_wmma_f32_16x16x32_bf16(
            false, a[r].v, false, bhi[j].v, (short)0, acc[r][j], false, false);
        acc[r][j] = __builtin_amdgcn_wmma_f32_16x16x32_bf16(
            false, a[r].v, false, blo[j].v, (short)0, acc[r][j], false, false);
      }
    }
  }

  // ---- epilogue: S += 0.5*Q and S += 0.5*Q^T (atomic), plus column sums ----
  const int m8 = 8 * half;
  const int nn = lane & 15;
#pragma unroll
  for (int r = 0; r < 2; ++r) {
#pragma unroll
    for (int j = 0; j < 4; ++j) {
#pragma unroll
      for (int v = 0; v < 8; ++v) {
        const int e = (2 * rp + r) * 16 + v + m8;  // C/D layout: M = v + 8*(lane>=16)
        const int t = (4 * chh + j) * 16 + nn;     // N = lane & 15
        const float val = 0.5f * acc[r][j][v];
        __hip_atomic_fetch_add(&S[e * CDIM + t], val,
                               __ATOMIC_RELAXED, __HIP_MEMORY_SCOPE_AGENT);
        __hip_atomic_fetch_add(&S[t * CDIM + e], val,
                               __ATOMIC_RELAXED, __HIP_MEMORY_SCOPE_AGENT);
      }
    }
  }
  __hip_atomic_fetch_add(&colsum[cg + 0], cs0, __ATOMIC_RELAXED, __HIP_MEMORY_SCOPE_AGENT);
  __hip_atomic_fetch_add(&colsum[cg + 1], cs1, __ATOMIC_RELAXED, __HIP_MEMORY_SCOPE_AGENT);
  __hip_atomic_fetch_add(&colsum[cg + 2], cs2, __ATOMIC_RELAXED, __HIP_MEMORY_SCOPE_AGENT);
  __hip_atomic_fetch_add(&colsum[cg + 3], cs3, __ATOMIC_RELAXED, __HIP_MEMORY_SCOPE_AGENT);
}

__global__ void zero_kernel(float* p, int n) {
  int i = blockIdx.x * 256 + threadIdx.x;
  if (i < n) p[i] = 0.f;
}

__global__ void __launch_bounds__(256)
finalize_kernel(const float* __restrict__ S, const float* __restrict__ colsum,
                const float* __restrict__ w_mu, const float* __restrict__ w_cov,
                const float* __restrict__ w_mu_out, const float* __restrict__ w_cov_out,
                const float* __restrict__ b_mu, const float* __restrict__ b_cov,
                float* __restrict__ out, float invN) {
  __shared__ float smu[CDIM];
  __shared__ float red[6 * 256];
  const int tid = threadIdx.x;
  if (tid < CDIM) smu[tid] = colsum[tid] * invN;
  __syncthreads();

  float c0 = 0.f, c1 = 0.f, c2 = 0.f;
  for (int k = 0; k < 64; ++k) {
    const int idx = tid + 256 * k;             // 128*128 / 256 = 64 per thread
    const int e = idx >> 7, t = idx & 127;
    const float cv = S[idx] * invN - smu[e] * smu[t];
    c0 += cv * w_cov[idx * 3 + 0];
    c1 += cv * w_cov[idx * 3 + 1];
    c2 += cv * w_cov[idx * 3 + 2];
  }
  float m0 = 0.f, m1 = 0.f, m2 = 0.f;
  if (tid < CDIM) {
    const float mv = smu[tid];
    m0 = mv * w_mu[tid * 3 + 0];
    m1 = mv * w_mu[tid * 3 + 1];
    m2 = mv * w_mu[tid * 3 + 2];
  }
  red[tid] = c0; red[256 + tid] = c1; red[512 + tid] = c2;
  red[768 + tid] = m0; red[1024 + tid] = m1; red[1280 + tid] = m2;
  __syncthreads();
  for (int s = 128; s > 0; s >>= 1) {
    if (tid < s) {
      red[tid] += red[tid + s];
      red[256 + tid] += red[256 + tid + s];
      red[512 + tid] += red[512 + tid + s];
      red[768 + tid] += red[768 + tid + s];
      red[1024 + tid] += red[1024 + tid + s];
      red[1280 + tid] += red[1280 + tid + s];
    }
    __syncthreads();
  }
  if (tid == 0) {
    float ch[3] = { red[0] + b_cov[0], red[256] + b_cov[1], red[512] + b_cov[2] };
    float mh[3] = { red[768] + b_mu[0], red[1024] + b_mu[1], red[1280] + b_mu[2] };
    float o = 0.f;
#pragma unroll
    for (int h = 0; h < HDIM; ++h) {
      const float aa = fmaxf(mh[h], 0.2f * mh[h]);   // leaky relu, LEAK = 0.2
      const float bb = fmaxf(ch[h], 0.2f * ch[h]);
      o += aa * w_mu_out[h] + bb * w_cov_out[h];
    }
    out[0] = o;
  }
}

extern "C" void kernel_launch(void* const* d_in, const int* in_sizes, int n_in,
                              void* d_out, int out_size, void* d_ws, size_t ws_size,
                              hipStream_t stream) {
  const float* x         = (const float*)d_in[0];
  const float* w_mu      = (const float*)d_in[1];
  const float* w_cov     = (const float*)d_in[2];
  const float* w_mu_out  = (const float*)d_in[3];
  const float* w_cov_out = (const float*)d_in[4];
  const float* b_mu      = (const float*)d_in[5];
  const float* b_cov     = (const float*)d_in[6];
  const int N = in_sizes[0] / CDIM;            // 1,000,000

  float* S      = (float*)d_ws;                // 128*128 fp32 Gram accumulator
  float* colsum = S + CDIM * CDIM;             // 128 fp32 column sums
  const int zn = CDIM * CDIM + CDIM;

  zero_kernel<<<(zn + 255) / 256, 256, 0, stream>>>(S, zn);
  gram_colsum_kernel<<<1024, 256, 0, stream>>>(x, S, colsum, N);
  finalize_kernel<<<1, 256, 0, stream>>>(S, colsum, w_mu, w_cov, w_mu_out, w_cov_out,
                                         b_mu, b_cov, (float*)d_out, 1.0f / (float)N);
}
